// RRB_91130616087312
// MI455X (gfx1250) — matrix operand order repaired and checked by reading, not compiled, not run on hardware
//
#include <hip/hip_runtime.h>
#include <hip/hip_fp16.h>

// MI455X / gfx1250: wave32, WMMA 16x16x32 f16 for the batched MLP GEMM chain.
// The reference's CNN branch (scene) is dead code and is skipped entirely.
// Weights are pre-packed once per call into WMMA B-fragment layout so the
// GEMM inner loop is pure b128 loads + v_wmma; the epilogue is fully
// compile-time specialized (LDY/relu/out-dtype) -> branch-free stores with
// immediate offsets.

typedef _Float16 h16;
typedef __attribute__((ext_vector_type(16))) _Float16 v16h;
typedef __attribute__((ext_vector_type(8)))  _Float16 v8h;
typedef __attribute__((ext_vector_type(8)))  float    v8f;

#define BATCH 4096
#define MTILES (BATCH / 16)

// ---------------------------------------------------------------------------
// Per-item prep: velocity features, neighbor selection, centerline scan,
// three small encoder MLPs -> 192-wide f16 feature row + f32 traj.
// One wave32 block per batch item.
// ---------------------------------------------------------------------------
struct EncW {
  const float* et_w0; const float* et_b0;
  const float* et_w1; const float* et_b1;
  const float* et_w2; const float* et_b2;
  const float* ev_w0; const float* ev_b0;
  const float* ev_w1; const float* ev_b1;
  const float* ev_w2; const float* ev_b2;
  const float* ie_w0; const float* ie_b0;
  const float* ie_w1; const float* ie_b1;
};

__global__ __launch_bounds__(32) void prep_kernel(
    const float* __restrict__ obs, const float* __restrict__ psv,
    const float* __restrict__ srv, const float* __restrict__ crg,
    EncW wp, h16* __restrict__ X0, float* __restrict__ trajb)
{
  const int b = blockIdx.x;
  const int lane = threadIdx.x;

  __shared__ float s_obs[320];            // obs[b]: (10,16,2)
  __shared__ float s_x[20];               // mlp input staging
  __shared__ float s_h0[32], s_h1[32];    // mlp hidden
  __shared__ float s_crx[256], s_cry[256], s_d[256];
  __shared__ float s_pts[20];
  __shared__ float s_bad[3];
  __shared__ int   s_ci[3];
  __shared__ float s_cd[3];

  const float* ob = obs + (size_t)b * 320;
  for (int i = lane; i < 320; i += 32) s_obs[i] = ob[i];
  const float scal = srv[b] / psv[b];
  __syncthreads();

  // ---- enc_traj input: v = diff of ego positions + [18,0], *sr/ps ----
  if (lane < 18) {
    int t = lane >> 1, c = lane & 1;
    float v = s_obs[((t + 1) * 16 + 0) * 2 + c] - s_obs[(t * 16 + 0) * 2 + c]
              + (c == 0 ? 18.f : 0.f);
    s_x[lane] = v * scal;
  }
  __syncthreads();
  { // 18 -> 32, relu
    float acc = wp.et_b0[lane];
    for (int i = 0; i < 18; ++i) acc += s_x[i] * wp.et_w0[i * 32 + lane];
    s_h0[lane] = fmaxf(acc, 0.f);
  }
  __syncthreads();
  { // 32 -> 32, relu
    float acc = wp.et_b1[lane];
    for (int i = 0; i < 32; ++i) acc += s_h0[i] * wp.et_w1[i * 32 + lane];
    s_h1[lane] = fmaxf(acc, 0.f);
  }
  __syncthreads();
  for (int j = lane; j < 64; j += 32) { // 32 -> 64, no relu
    float acc = wp.et_b2[j];
    for (int i = 0; i < 32; ++i) acc += s_h1[i] * wp.et_w2[i * 64 + j];
    X0[(size_t)b * 192 + j] = (h16)acc;
  }
  __syncthreads();

  // ---- neighbor selection ----
  if (lane < 16) {
    float ex_raw = s_obs[(9 * 16 + 0) * 2 + 0];
    float ax = s_obs[(9 * 16 + lane) * 2 + 0];
    float ay = s_obs[(9 * 16 + lane) * 2 + 1];
    bool behind = ax > ex_raw;                 // behind[0] == false
    float cx = behind ? -1000.f : ax; if (cx != cx) cx = -1000.f;
    float cy = behind ? -1000.f : ay; if (cy != cy) cy = -1000.f;
    float e0x = s_obs[(9 * 16 + 0) * 2 + 0]; if (e0x != e0x) e0x = -1000.f;
    float e0y = s_obs[(9 * 16 + 0) * 2 + 1]; if (e0y != e0y) e0y = -1000.f;
    float dx = cx - e0x, dy = cy - e0y;
    s_d[lane] = dx * dx + dy * dy;
  }
  __syncthreads();
  if (lane == 0) {  // 4 smallest, ties -> lowest index; skip the first (self)
    unsigned sel = 0;
    for (int k = 0; k < 4; ++k) {
      float bv = 3.4e38f; int bi = 0;
      for (int n = 0; n < 16; ++n) {
        if (!((sel >> n) & 1u) && s_d[n] < bv) { bv = s_d[n]; bi = n; }
      }
      sel |= 1u << bi;
      if (k > 0) { s_ci[k - 1] = bi; s_cd[k - 1] = bv; }
    }
  }
  __syncthreads();
  // rel[t'][k][c] = ego_last - obs[-3+t'][ci[k]]   (raw obs for the gather)
  if (lane < 18) {
    int tp = lane / 6, rem = lane % 6, k = rem >> 1, c = rem & 1;
    float nbv = s_obs[((7 + tp) * 16 + s_ci[k]) * 2 + c];
    s_x[lane] = s_obs[(9 * 16 + 0) * 2 + c] - nbv;
  }
  __syncthreads();
  if (lane < 3) {
    float s = 0.f;
    for (int tp = 0; tp < 3; ++tp)
      for (int c = 0; c < 2; ++c) s += s_x[tp * 6 + lane * 2 + c];
    s_bad[lane] = ((s_cd[lane] > 600000.f) || (s != s)) ? 1.f : 0.f;
  }
  __syncthreads();
  if (lane < 18) {
    int rem = lane % 6, k = rem >> 1, c = rem & 1;
    float v = (s_bad[k] != 0.f) ? 2000.f : s_x[lane];
    s_x[lane] = v * scal - (c == 0 ? 10.f : 7.f);
  }
  __syncthreads();
  { // ev MLP 18->32->32->64 (same shape as et)
    float acc = wp.ev_b0[lane];
    for (int i = 0; i < 18; ++i) acc += s_x[i] * wp.ev_w0[i * 32 + lane];
    s_h0[lane] = fmaxf(acc, 0.f);
  }
  __syncthreads();
  {
    float acc = wp.ev_b1[lane];
    for (int i = 0; i < 32; ++i) acc += s_h0[i] * wp.ev_w1[i * 32 + lane];
    s_h1[lane] = fmaxf(acc, 0.f);
  }
  __syncthreads();
  for (int j = lane; j < 64; j += 32) {
    float acc = wp.ev_b2[j];
    for (int i = 0; i < 32; ++i) acc += s_h1[i] * wp.ev_w2[i * 64 + j];
    X0[(size_t)b * 192 + 64 + j] = (h16)acc;
  }

  // ---- centerline scan ----
  float lpx = s_obs[(9 * 16 + 0) * 2 + 0], lpy = s_obs[(9 * 16 + 0) * 2 + 1];
  float lvx = lpx - s_obs[(8 * 16 + 0) * 2 + 0];
  float lvy = lpy - s_obs[(8 * 16 + 0) * 2 + 1];
  float speed = sqrtf(lvx * lvx + lvy * lvy);
  const float* crp = crg + (size_t)b * 512;
  __syncthreads();
  for (int j = lane; j < 256; j += 32) {
    float x = crp[j * 2 + 0], y = crp[j * 2 + 1];
    s_crx[j] = x; s_cry[j] = y;
    float dx = x - lpx, dy = y - lpy;
    s_d[j] = sqrtf(dx * dx + dy * dy);
  }
  __syncthreads();
  float qerr = 0.f, prevx = lpx, prevy = lpy;
  for (int t = 0; t < 10; ++t) {
    float target = speed - qerr;
    float bv = 3.4e38f; int bi = 0x7fffffff;
    for (int j = lane; j < 256; j += 32) {
      float v = fabsf(s_d[j] - target);
      if (v < bv || (v == bv && j < bi)) { bv = v; bi = j; }
    }
    for (int off = 16; off; off >>= 1) {   // wave argmin, ties -> lowest idx
      float ov = __shfl_xor(bv, off, 32);
      int   oi = __shfl_xor(bi, off, 32);
      if (ov < bv || (ov == bv && oi < bi)) { bv = ov; bi = oi; }
    }
    float ptx = s_crx[bi], pty = s_cry[bi];
    __syncthreads();
    for (int j = lane; j < bi; j += 32) { s_crx[j] = 50000.f; s_cry[j] = 50000.f; }
    __syncthreads();
    for (int j = lane; j < 256; j += 32) {
      float dx = s_crx[j] - ptx, dy = s_cry[j] - pty;
      s_d[j] = sqrtf(dx * dx + dy * dy);
    }
    float mx = ptx - prevx, my = pty - prevy;
    float mapped = sqrtf(mx * mx + my * my);
    qerr = (t > 0) ? (mapped - speed) : 0.f;
    prevx = ptx; prevy = pty;
    if (lane == 0) { s_pts[2 * t] = ptx; s_pts[2 * t + 1] = pty; }
    __syncthreads();
  }

  // ---- traj offset + kd input ----
  float offs = s_obs[(9 * 16 + 0) * 2 + 1] - s_pts[1];
  if (lane < 20) {
    int c = lane & 1;
    float tr = (c == 0) ? s_pts[lane] : (s_pts[lane] + offs * 0.75f);
    trajb[(size_t)b * 20 + lane] = tr;
    s_x[lane] = tr - s_obs[(9 * 16 + 0) * 2 + c] + (c == 0 ? 70.f : 0.5f);
  }
  __syncthreads();
  { // ie: 20 -> 32, relu
    float acc = wp.ie_b0[lane];
    for (int i = 0; i < 20; ++i) acc += s_x[i] * wp.ie_w0[i * 32 + lane];
    s_h0[lane] = fmaxf(acc, 0.f);
  }
  __syncthreads();
  for (int j = lane; j < 64; j += 32) { // 32 -> 64, relu (final_relu=True)
    float acc = wp.ie_b1[j];
    for (int i = 0; i < 32; ++i) acc += s_h0[i] * wp.ie_w1[i * 64 + j];
    X0[(size_t)b * 192 + 128 + j] = (h16)fmaxf(acc, 0.f);
  }
}

// ---------------------------------------------------------------------------
// Weight pre-pack: f32 W(KxN) -> f16 WMMA B fragments, one 32B run per lane
// per (n-tile, k-tile).  Also zero-pads the bias to Ntiles*16.
// Fragment layout: frag[(((nt*Ktiles)+kt)*32 + lane)*16 + e], where
//   col = nt*16 + (lane&15), half = lane>>4, K-index = kt*32 + half*16 + e.
// ---------------------------------------------------------------------------
__global__ void pack_weights(const float* __restrict__ W,
                             const float* __restrict__ bias,
                             int K, int N, int Ntiles,
                             h16* __restrict__ frag, float* __restrict__ biasPad)
{
  int tid = blockIdx.x * blockDim.x + threadIdx.x;
  int Ktiles = K >> 5;
  int total = Ntiles * Ktiles * 32;
  if (tid < Ntiles * 16) {
    biasPad[tid] = (tid < N) ? bias[tid] : 0.f;
  }
  if (tid >= total) return;
  int lane = tid & 31;
  int t    = tid >> 5;
  int kt   = t % Ktiles;      // tiny grids; div fine here
  int nt   = t / Ktiles;
  int col  = nt * 16 + (lane & 15);
  int half = lane >> 4;
  h16* out = frag + tid * 16;
  bool ok = (col < N);
#pragma unroll
  for (int e = 0; e < 16; ++e) {
    int kk = kt * 32 + half * 16 + e;
    float w = ok ? W[kk * N + col] : 0.f;
    out[e] = (h16)w;
  }
}

// ---------------------------------------------------------------------------
// WMMA GEMM: Y = act(X(f16, 4096xK) @ Wfrag + biasPad).  One wave per 16x16
// tile.  Everything layer-specific is compile-time: the K loop fully unrolls
// into b128-load + v_wmma chains, and the epilogue is one address compute
// plus 8 stores with immediate offsets (no branches, no cndmask).
// ---------------------------------------------------------------------------
template<int K, int NT, int LDY, bool RELU, bool F32OUT>
__global__ __launch_bounds__(256) void gemm_wmma(
    const h16* __restrict__ X, const h16* __restrict__ frag,
    const float* __restrict__ biasPad, void* __restrict__ Y)
{
  constexpr int KT = K / 32;
  int wid  = (int)((blockIdx.x * blockDim.x + threadIdx.x) >> 5);
  int lane = threadIdx.x & 31;
  int mt = wid / NT;
  if (mt >= MTILES) return;                 // wave-uniform: EXEC stays all-1s
  int nt = wid - mt * NT;
  int m0 = mt * 16, n0 = nt * 16;
  int row  = m0 + (lane & 15);
  int half = lane >> 4;
  int col  = n0 + (lane & 15);

  v8f acc = {0.f, 0.f, 0.f, 0.f, 0.f, 0.f, 0.f, 0.f};
  const h16* xrow = X + row * K;
  const h16* bp   = frag + (nt * KT) * 512 + lane * 16;   // 512 f16 per k-tile

#pragma unroll
  for (int kt = 0; kt < KT; ++kt) {
    // A 16x32 f16 fragment: two contiguous 16B runs per lane
    v8h alo = *(const v8h*)(xrow + kt * 32 + half * 16);
    v8h ahi = *(const v8h*)(xrow + kt * 32 + half * 16 + 8);
    // B 32x16 f16 fragment: one contiguous 32B run per lane (pre-packed)
    v8h blo = *(const v8h*)(bp + kt * 512);
    v8h bhi = *(const v8h*)(bp + kt * 512 + 8);
    v16h a{}, bm{};
#pragma unroll
    for (int e = 0; e < 8; ++e) {
      a[e] = alo[e];  a[e + 8] = ahi[e];
      bm[e] = blo[e]; bm[e + 8] = bhi[e];
    }
    acc = __builtin_amdgcn_wmma_f32_16x16x32_f16(
        false, a, false, bm, (short)0, acc, false, false);
  }

  float bb = biasPad[col];
  int ybase = (m0 + 8 * half) * LDY + col;  // C/D: VGPR r -> M = r + 8*half
#pragma unroll
  for (int r = 0; r < 8; ++r) {
    float v = acc[r] + bb;
    if (RELU) v = fmaxf(v, 0.f);
    if (F32OUT) ((float*)Y)[ybase + r * LDY] = v;
    else        ((h16*)Y)[ybase + r * LDY] = (h16)v;
  }
}

// ---------------------------------------------------------------------------
// Final assembly: out[b,0,t,:] = [traj+tanh(rm)*33, exp(rv0), exp(rv1), tanh(rv2)]
// rm has padded stride 32, rv padded stride 80.
// ---------------------------------------------------------------------------
__global__ void finalize_kernel(const float* __restrict__ traj,
                                const float* __restrict__ rm,
                                const float* __restrict__ rv,
                                float* __restrict__ out)
{
  int i = blockIdx.x * blockDim.x + threadIdx.x;
  if (i >= BATCH * 10) return;
  int b = i / 10, t = i % 10;
  float tx = traj[b * 20 + t * 2 + 0];
  float ty = traj[b * 20 + t * 2 + 1];
  float mx = tanhf(rm[b * 32 + t * 2 + 0]);
  float my = tanhf(rm[b * 32 + t * 2 + 1]);
  const float* v = rv + b * 80 + t * 7;
  float* o = out + b * 50 + t * 5;
  o[0] = tx + mx * 33.f;
  o[1] = ty + my * 33.f;
  o[2] = expf(v[0]);
  o[3] = expf(v[1]);
  o[4] = tanhf(v[2]);
}

// ---------------------------------------------------------------------------
// Host launcher
// ---------------------------------------------------------------------------
extern "C" void kernel_launch(void* const* d_in, const int* in_sizes, int n_in,
                              void* d_out, int out_size, void* d_ws, size_t ws_size,
                              hipStream_t stream)
{
  (void)out_size; (void)ws_size;
  const float* obs = (const float*)d_in[0];
  const float* ps  = (const float*)d_in[1];
  const float* sr  = (const float*)d_in[2];
  // d_in[3] = scene: dead code in the reference, never read
  const float* cr  = (const float*)d_in[4];

  // Locate param leaves by element-count; biases follow their weights.
  auto find_nth = [&](int want, int occur) -> int {
    int c = 0;
    for (int i = 5; i < n_in; ++i)
      if (in_sizes[i] == want && ++c == occur) return i;
    return -1;
  };
  auto P = [&](int idx, int fb) -> const float* {
    return (const float*)d_in[idx >= 0 ? idx : fb];
  };

  int i_et0 = find_nth(576, 1),   i_ev0 = find_nth(576, 2);
  int i_et1 = find_nth(1024, 1),  i_ev1 = find_nth(1024, 2);
  int i_et2 = find_nth(2048, 1),  i_ev2 = find_nth(2048, 2), i_ie1 = find_nth(2048, 3);
  int i_ie0 = find_nth(640, 1);
  int i_d0  = find_nth(49152, 1), i_d1 = find_nth(32768, 1);
  int i_d2  = find_nth(16384, 1), i_d3 = find_nth(16384, 2);
  int i_rm0 = find_nth(8192, 1),  i_rm1 = find_nth(1280, 1);
  int i_rv0 = find_nth(8192, 2),  i_rv1 = find_nth(4480, 1);

  EncW wp;
  wp.et_w0 = P(i_et0, 19); wp.et_b0 = P(i_et0 + 1, 20);
  wp.et_w1 = P(i_et1, 21); wp.et_b1 = P(i_et1 + 1, 22);
  wp.et_w2 = P(i_et2, 23); wp.et_b2 = P(i_et2 + 1, 24);
  wp.ev_w0 = P(i_ev0, 25); wp.ev_b0 = P(i_ev0 + 1, 26);
  wp.ev_w1 = P(i_ev1, 27); wp.ev_b1 = P(i_ev1 + 1, 28);
  wp.ev_w2 = P(i_ev2, 29); wp.ev_b2 = P(i_ev2 + 1, 30);
  wp.ie_w0 = P(i_ie0, 31); wp.ie_b0 = P(i_ie0 + 1, 32);
  wp.ie_w1 = P(i_ie1, 33); wp.ie_b1 = P(i_ie1 + 1, 34);

  const float* dec_w0 = P(i_d0, 11),  *dec_b0 = P(i_d0 + 1, 12);
  const float* dec_w1 = P(i_d1, 13),  *dec_b1 = P(i_d1 + 1, 14);
  const float* dec_w2 = P(i_d2, 15),  *dec_b2 = P(i_d2 + 1, 16);
  const float* dc2_w0 = P(i_d3, 17),  *dc2_b0 = P(i_d3 + 1, 18);
  const float* rm_w0  = P(i_rm0, 35), *rm_b0  = P(i_rm0 + 1, 36);
  const float* rm_w1  = P(i_rm1, 37), *rm_b1  = P(i_rm1 + 1, 38);
  const float* rv_w0  = P(i_rv0, 39), *rv_b0  = P(i_rv0 + 1, 40);
  const float* rv_w1  = P(i_rv1, 41), *rv_b1  = P(i_rv1 + 1, 42);

  // Workspace layout (256B aligned chunks)
  char* base = (char*)d_ws;
  size_t off = 0;
  auto alloc = [&](size_t bytes) -> char* {
    char* p = base + off;
    off = (off + bytes + 255) & ~(size_t)255;
    return p;
  };
  h16*   X0    = (h16*)  alloc((size_t)BATCH * 192 * sizeof(h16));
  float* trajf = (float*)alloc((size_t)BATCH * 20 * sizeof(float));
  h16*   a1    = (h16*)  alloc((size_t)BATCH * 256 * sizeof(h16));
  h16*   a2    = (h16*)  alloc((size_t)BATCH * 128 * sizeof(h16));
  h16*   a3    = (h16*)  alloc((size_t)BATCH * 128 * sizeof(h16));
  h16*   feat  = (h16*)  alloc((size_t)BATCH * 128 * sizeof(h16));
  h16*   rmh   = (h16*)  alloc((size_t)BATCH * 64 * sizeof(h16));
  h16*   rvh   = (h16*)  alloc((size_t)BATCH * 64 * sizeof(h16));
  float* rmf   = (float*)alloc((size_t)BATCH * 32 * sizeof(float));   // N=20 pad 32
  float* rvf   = (float*)alloc((size_t)BATCH * 80 * sizeof(float));   // N=70 pad 80

  // Packed weight fragments + padded biases (f16 frags: Ntiles*Ktiles*512 elems)
  struct Layer { const float *W, *b; int K, N, NT; h16* frag; float* bpad; };
  Layer L[8] = {
    { dec_w0, dec_b0, 192, 256, 16, nullptr, nullptr },
    { dec_w1, dec_b1, 256, 128,  8, nullptr, nullptr },
    { dec_w2, dec_b2, 128, 128,  8, nullptr, nullptr },
    { dc2_w0, dc2_b0, 128, 128,  8, nullptr, nullptr },
    { rm_w0,  rm_b0,  128,  64,  4, nullptr, nullptr },
    { rm_w1,  rm_b1,   64,  20,  2, nullptr, nullptr },
    { rv_w0,  rv_b0,  128,  64,  4, nullptr, nullptr },
    { rv_w1,  rv_b1,   64,  70,  5, nullptr, nullptr },
  };
  for (int l = 0; l < 8; ++l) {
    int KT = L[l].K / 32;
    L[l].frag = (h16*)  alloc((size_t)L[l].NT * KT * 512 * sizeof(h16));
    L[l].bpad = (float*)alloc((size_t)L[l].NT * 16 * sizeof(float));
  }

  // 0) pack weights (tiny, once per call)
  for (int l = 0; l < 8; ++l) {
    int KT = L[l].K / 32;
    int total = L[l].NT * KT * 32;
    int blocks = (total + 127) / 128;
    pack_weights<<<blocks, 128, 0, stream>>>(L[l].W, L[l].b, L[l].K, L[l].N,
                                             L[l].NT, L[l].frag, L[l].bpad);
  }

  // 1) prep + encoders + scan
  prep_kernel<<<BATCH, 32, 0, stream>>>(obs, ps, sr, cr, wp, X0, trajf);

  // 2+3) GEMM chain + heads (template args: K, NT, LDY, RELU, F32OUT)
  auto blocks_for = [](int NT) { return (MTILES * NT + 7) / 8; };
  gemm_wmma<192,16,256,true ,false><<<blocks_for(16), 256, 0, stream>>>(X0,   L[0].frag, L[0].bpad, a1);
  gemm_wmma<256, 8,128,true ,false><<<blocks_for( 8), 256, 0, stream>>>(a1,   L[1].frag, L[1].bpad, a2);
  gemm_wmma<128, 8,128,true ,false><<<blocks_for( 8), 256, 0, stream>>>(a2,   L[2].frag, L[2].bpad, a3);
  gemm_wmma<128, 8,128,true ,false><<<blocks_for( 8), 256, 0, stream>>>(a3,   L[3].frag, L[3].bpad, feat);
  gemm_wmma<128, 4, 64,true ,false><<<blocks_for( 4), 256, 0, stream>>>(feat, L[4].frag, L[4].bpad, rmh);
  gemm_wmma< 64, 2, 32,false,true ><<<blocks_for( 2), 256, 0, stream>>>(rmh,  L[5].frag, L[5].bpad, rmf);
  gemm_wmma<128, 4, 64,true ,false><<<blocks_for( 4), 256, 0, stream>>>(feat, L[6].frag, L[6].bpad, rvh);
  gemm_wmma< 64, 5, 80,false,true ><<<blocks_for( 5), 256, 0, stream>>>(rvh,  L[7].frag, L[7].bpad, rvf);

  // 4) assemble output
  finalize_kernel<<<(BATCH * 10 + 255) / 256, 256, 0, stream>>>(
      trajf, rmf, rvf, (float*)d_out);
}
